// ALL_9320079032780
// MI455X (gfx1250) — compile-verified, matrix operand order checked by reading, not run on hardware
//
#include <hip/hip_runtime.h>
#include <hip/hip_bf16.h>

// ---------------------------------------------------------------------------
// CPC InfoNCE loss for MI455X (gfx1250), bf16 WMMA path.
//   N=256, C=128, P=275 (11*25), D=128, DM=DC=64
// ---------------------------------------------------------------------------

typedef __bf16 bf16;
typedef __attribute__((ext_vector_type(16))) __bf16 v16bf;
typedef __attribute__((ext_vector_type(8)))  __bf16 bf16x8;
typedef __attribute__((ext_vector_type(8)))  float  v8f;
typedef __attribute__((ext_vector_type(4)))  float  f32x4;

#define NN   256
#define CC   128
#define DD   128
#define PP   275

__device__ __forceinline__ v8f wmma_bf16(v16bf a, v16bf b, v8f c) {
  // D = A(16x32 bf16) * B(32x16 bf16) + C(16x16 f32)
  return __builtin_amdgcn_wmma_f32_16x16x32_bf16(
      /*neg_a=*/false, a, /*neg_b=*/false, b,
      /*c_mod=*/(short)0, c, /*reuse_a=*/false, /*reuse_b=*/false);
}

// Gather one wave-lane's A/B fragment from a row-major [row][K] bf16 buffer.
// ISA 16-bit operand layout: lanes 0-15 hold K 0-7 (regs 0-3) and 16-23
// (regs 4-7); lanes 16-31 hold K 8-15 and 24-31. Two 16B chunks per lane.
__device__ __forceinline__ v16bf load_frag(const bf16* __restrict__ row,
                                           int k0, int lane) {
  const int lo = (lane & 16) ? 8 : 0;
  union { v16bf v; bf16x8 h[2]; } u;
  u.h[0] = *reinterpret_cast<const bf16x8*>(row + k0 + lo);
  u.h[1] = *reinterpret_cast<const bf16x8*>(row + k0 + 16 + lo);
  return u.v;
}

// --------------------------------------------------------------------------
// bias_t / bias_p = m @ Wm + c @ Wc + b   (fp32, tiny), stored TRANSPOSED
// as (d, n) so the main kernel's per-lane bias reads (fixed d, 8 consecutive
// n) are two dense b128 loads instead of eight strided b32 gathers.
// grid: 256 (n), block: 128 (d)
// --------------------------------------------------------------------------
__global__ void cpc_bias(const float* __restrict__ mt, const float* __restrict__ mp,
                         const float* __restrict__ ct, const float* __restrict__ cp,
                         const float* __restrict__ W,  const float* __restrict__ b,
                         float* __restrict__ bTt, float* __restrict__ bPt) {
  const int n = blockIdx.x, d = threadIdx.x;
  float aT = b[d], aP = b[d];
  for (int k = 0; k < 64; ++k) {
    const float w = W[(128 + k) * DD + d];            // Wm rows
    aT = fmaf(mt[n * 64 + k], w, aT);
    aP = fmaf(mp[n * 64 + k], w, aP);
  }
  for (int k = 0; k < 64; ++k) {
    const float w = W[(192 + k) * DD + d];            // Wc rows
    aT = fmaf(ct[n * 64 + k], w, aT);
    aP = fmaf(cp[n * 64 + k], w, aP);
  }
  bTt[d * NN + n] = aT;    // transposed layout (d, n)
  bPt[d * NN + n] = aP;
}

// WlT[d][c] = (bf16) W_join[c][d]   (row per output column -> contiguous B loads)
__global__ void cpc_wlt(const float* __restrict__ W, bf16* __restrict__ WlT) {
  const int d = blockIdx.x, c = threadIdx.x;
  WlT[d * CC + c] = (bf16)W[c * DD + d];
}

__global__ void cpc_fbf(const float* __restrict__ f, bf16* __restrict__ fB) {
  const int i = blockIdx.x * 256 + threadIdx.x;       // 128 * 256 = 32768
  fB[i] = (bf16)f[i];
}

// --------------------------------------------------------------------------
// (N,C,P) f32  ->  (P,N,C) bf16, LDS-tiled transpose, coalesced both sides.
// grid: (ceil(P/32)=9, 4 c-tiles * 2 tensors = 8, N=256), block (32,8)
// --------------------------------------------------------------------------
__global__ void cpc_transpose(const float* __restrict__ xt, const float* __restrict__ xp,
                              bf16* __restrict__ XtB, bf16* __restrict__ XpB) {
  __shared__ float tile[32][33];
  const int pt = blockIdx.x;
  const int ctile = blockIdx.y & 3;
  const int which = blockIdx.y >> 2;
  const int n  = blockIdx.z;
  const float* src = which ? xp  : xt;
  bf16*        dst = which ? XpB : XtB;
  const int tx = threadIdx.x, ty = threadIdx.y;
  const int p0 = pt * 32, c0 = ctile * 32;
#pragma unroll
  for (int i = 0; i < 4; ++i) {
    const int c  = c0 + ty + i * 8;
    const int pp = p0 + tx;
    tile[ty + i * 8][tx] = (pp < PP) ? src[((size_t)n * CC + c) * PP + pp] : 0.0f;
  }
  __syncthreads();
#pragma unroll
  for (int i = 0; i < 4; ++i) {
    const int pp = p0 + ty + i * 8;
    const int c  = c0 + tx;
    if (pp < PP)
      dst[((size_t)pp * NN + n) * CC + c] = (bf16)tile[tx][ty + i * 8];
  }
}

// --------------------------------------------------------------------------
// Main kernel: one block per p.  512 threads = 16 waves.
// Stage A: positive/pred = X @ Wl + bias  -> bf16 tiles in LDS.
// Stage B: logits1 = f @ positive^T, logits2 = pred @ positive^T with online
//          row-wise logsumexp and diagonal capture; fixed-order reduction.
// Dynamic LDS: 2 * 256*128 bf16 (128 KB) + 16 wave partials
// -> 2 workgroups per 320 KB WGP.
// --------------------------------------------------------------------------
__global__ __launch_bounds__(512) void cpc_main(
    const bf16* __restrict__ XtB, const bf16* __restrict__ XpB,
    const bf16* __restrict__ WlT, const bf16* __restrict__ fB,
    const float* __restrict__ biasTt, const float* __restrict__ biasPt,
    float* __restrict__ partial) {
  extern __shared__ char smem[];
  bf16*  sPos  = (bf16*)smem;                       // [256][128]
  bf16*  sPred = sPos + NN * DD;                    // [256][128]
  float* wsum  = (float*)(smem + 2 * NN * DD * sizeof(bf16));

  const int p    = blockIdx.x;
  const int tid  = threadIdx.x;
  const int w    = tid >> 5;          // wave id 0..15
  const int lane = tid & 31;
  const int l15  = lane & 15;
  const int half = (lane >> 4) & 1;   // C/D rows: M = r + 8*half

  // ---------------- Stage A: positive & pred -> LDS bf16 ----------------
  const int arow = w * 16 + l15;                      // A-fragment row
  const bf16* xpRow = XpB + ((size_t)p * NN + arow) * CC;
  const bf16* xtRow = XtB + ((size_t)p * NN + arow) * CC;
  v16bf ap[4], at[4];
#pragma unroll
  for (int k = 0; k < 4; ++k) {
    ap[k] = load_frag(xpRow, k * 32, lane);
    at[k] = load_frag(xtRow, k * 32, lane);
  }

  const int gm0 = w * 16 + 8 * half;   // first of this lane's 8 output rows

#pragma unroll 1
  for (int dt = 0; dt < 8; ++dt) {
    const bf16* wcol = WlT + (size_t)(dt * 16 + l15) * CC;   // B col = WlT row
    v8f accP = {0, 0, 0, 0, 0, 0, 0, 0};
    v8f accT = {0, 0, 0, 0, 0, 0, 0, 0};
#pragma unroll
    for (int k = 0; k < 4; ++k) {
      const v16bf bfr = load_frag(wcol, k * 32, lane);
      accP = wmma_bf16(ap[k], bfr, accP);
      accT = wmma_bf16(at[k], bfr, accT);
    }
    const int gd = dt * 16 + l15;
    // bias is stored (d, n): this lane's 8 rows are contiguous -> 2x b128 each
    const f32x4* bPv = reinterpret_cast<const f32x4*>(biasPt + (size_t)gd * NN + gm0);
    const f32x4* bTv = reinterpret_cast<const f32x4*>(biasTt + (size_t)gd * NN + gm0);
    const f32x4 bp0 = bPv[0], bp1 = bPv[1];
    const f32x4 bt0 = bTv[0], bt1 = bTv[1];
#pragma unroll
    for (int r = 0; r < 4; ++r) {
      sPos [(gm0 + r    ) * DD + gd] = (bf16)(accP[r]     + bp0[r]);
      sPos [(gm0 + r + 4) * DD + gd] = (bf16)(accP[r + 4] + bp1[r]);
      sPred[(gm0 + r    ) * DD + gd] = (bf16)(accT[r]     + bt0[r]);
      sPred[(gm0 + r + 4) * DD + gd] = (bf16)(accT[r + 4] + bt1[r]);
    }
  }
  __syncthreads();

  // ---------------- Stage B: logits + online logsumexp ----------------
  const bf16* fRow = fB + (size_t)arow * DD;
  const bf16* pRow = sPred + arow * DD;
  v16bf a1[4], a2[4];
#pragma unroll
  for (int k = 0; k < 4; ++k) {
    a1[k] = load_frag(fRow, k * 32, lane);   // logits1 A = f
    a2[k] = load_frag(pRow, k * 32, lane);   // logits2 A = pred
  }

  float mx1[8], s1[8], mx2[8], s2[8];
#pragma unroll
  for (int r = 0; r < 8; ++r) { mx1[r] = -1e30f; s1[r] = 0.f; mx2[r] = -1e30f; s2[r] = 0.f; }
  float dg = 0.f;   // diagonal accumulator (logits1 + logits2)

#pragma unroll 1
  for (int mt = 0; mt < 16; ++mt) {
    v8f c1 = {0, 0, 0, 0, 0, 0, 0, 0};
    v8f c2 = {0, 0, 0, 0, 0, 0, 0, 0};
    const bf16* bcol = sPos + (size_t)(mt * 16 + l15) * DD;   // B col = positive row
#pragma unroll
    for (int k = 0; k < 4; ++k) {
      const v16bf bfr = load_frag(bcol, k * 32, lane);        // shared by both GEMMs
      c1 = wmma_bf16(a1[k], bfr, c1);
      c2 = wmma_bf16(a2[k], bfr, c2);
    }
    const bool dtile = (mt == w);
#pragma unroll
    for (int r = 0; r < 8; ++r) {
      // diagonal: M==N lives on lane r (rows 0-7) and lane r+24 (rows 8-15)
      if (dtile && ((lane == r) || (lane == r + 24))) dg += c1[r] + c2[r];
      float t1 = c1[r], t2 = c2[r];
      float m1 = t1, m2 = t2;
#pragma unroll
      for (int s = 1; s < 16; s <<= 1) {
        m1 = fmaxf(m1, __shfl_xor(m1, s, 16));
        m2 = fmaxf(m2, __shfl_xor(m2, s, 16));
      }
      const float n1 = fmaxf(mx1[r], m1);
      const float n2 = fmaxf(mx2[r], m2);
      float e1 = __expf(t1 - n1);
      float e2 = __expf(t2 - n2);
#pragma unroll
      for (int s = 1; s < 16; s <<= 1) {
        e1 += __shfl_xor(e1, s, 16);
        e2 += __shfl_xor(e2, s, 16);
      }
      s1[r] = s1[r] * __expf(mx1[r] - n1) + e1;  mx1[r] = n1;
      s2[r] = s2[r] * __expf(mx2[r] - n2) + e2;  mx2[r] = n2;
    }
  }

  // per-wave contribution: sum_rows (lse1 + lse2) - sum diag
  float v = -dg;
  if (lane == 0 || lane == 16) {   // lane 0: rows 0-7, lane 16: rows 8-15
#pragma unroll
    for (int r = 0; r < 8; ++r)
      v += mx1[r] + __logf(s1[r]) + mx2[r] + __logf(s2[r]);
  }
#pragma unroll
  for (int s = 1; s < 32; s <<= 1) v += __shfl_xor(v, s, 32);
  if (lane == 0) wsum[w] = v;
  __syncthreads();
  if (tid == 0) {
    float t = 0.f;
    for (int i = 0; i < 16; ++i) t += wsum[i];   // fixed order: deterministic
    partial[p] = t;
  }
}

__global__ void cpc_finalize(const float* __restrict__ partial, float* __restrict__ out) {
  if (blockIdx.x == 0 && threadIdx.x == 0) {
    float t = 0.f;
    for (int i = 0; i < PP; ++i) t += partial[i];  // fixed order
    out[0] = t * (1.0f / (float(PP) * float(NN)));
  }
}

// --------------------------------------------------------------------------
extern "C" void kernel_launch(void* const* d_in, const int* in_sizes, int n_in,
                              void* d_out, int out_size, void* d_ws, size_t ws_size,
                              hipStream_t stream) {
  (void)in_sizes; (void)n_in; (void)out_size; (void)ws_size;
  const float* f  = (const float*)d_in[0];   // (256,128)
  const float* xt = (const float*)d_in[1];   // (256,128,11,25)
  const float* xp = (const float*)d_in[2];
  const float* mt = (const float*)d_in[3];   // (256,64)
  const float* mp = (const float*)d_in[4];
  const float* ct = (const float*)d_in[5];
  const float* cp = (const float*)d_in[6];
  const float* W  = (const float*)d_in[7];   // (256,128)
  const float* b  = (const float*)d_in[8];   // (128,)

  char* ws = (char*)d_ws;
  bf16*  WlT     = (bf16*) (ws + 0);                       //  32 KB
  bf16*  fB      = (bf16*) (ws + 32768);                   //  64 KB
  float* bTt     = (float*)(ws + 98304);                   // 128 KB, (d,n)
  float* bPt     = (float*)(ws + 229376);                  // 128 KB, (d,n)
  float* partial = (float*)(ws + 360448);                  // 275 * 4 B
  bf16*  XtB     = (bf16*) (ws + 364544);                  // 17.2 MB
  bf16*  XpB     = (bf16*) (ws + 364544 + 18022400);       // 17.2 MB

  cpc_bias<<<NN, DD, 0, stream>>>(mt, mp, ct, cp, W, b, bTt, bPt);
  cpc_wlt<<<DD, CC, 0, stream>>>(W, WlT);
  cpc_fbf<<<128, 256, 0, stream>>>(f, fB);
  cpc_transpose<<<dim3(9, 8, NN), dim3(32, 8), 0, stream>>>(xt, xp, XtB, XpB);

  const size_t ldsBytes = 2 * NN * DD * sizeof(bf16) + 64 * sizeof(float);
  cpc_main<<<PP, 512, ldsBytes, stream>>>(XtB, XpB, WlT, fB, bTt, bPt, partial);
  cpc_finalize<<<1, 32, 0, stream>>>(partial, (float*)d_out);
}
